// PointNet2Contrastive_78254304133663
// MI455X (gfx1250) — matrix operand order repaired and checked by reading, not compiled, not run on hardware
//
#include <hip/hip_runtime.h>
#include <hip/hip_bf16.h>
#include <math.h>

typedef __attribute__((ext_vector_type(16))) _Float16 v16h;
typedef __attribute__((ext_vector_type(8)))  _Float16 v8h;
typedef __attribute__((ext_vector_type(8)))  float    v8f;

#define BN_SCALE 0.9999950000374997f

// ---------------------------------------------------------------------------
// CDNA5 async global->LDS copy (ASYNCcnt) helpers.
// lds_off: byte offset within wave's LDS (low 32 bits of the flat LDS address).
// ---------------------------------------------------------------------------
__device__ __forceinline__ void async_load_lds_b128(unsigned lds_off, const void* gptr) {
  asm volatile("global_load_async_to_lds_b128 %0, %1, off"
               :: "v"(lds_off), "v"(gptr) : "memory");
}
__device__ __forceinline__ void wait_asynccnt0() {
  asm volatile("s_wait_asynccnt 0x0" ::: "memory");
}

// ---------------------------------------------------------------------------
// Weight folding: Wf16[o][k] = W[o][k] * (g[o]*BN_SCALE), K zero-padded to 32.
// bias_f = scale*lin_b + bn_b  (any of g / lin_b / bn_b may be null)
// ---------------------------------------------------------------------------
__global__ void fold_weights_kernel(const float* __restrict__ W,
                                    const float* __restrict__ g,
                                    const float* __restrict__ lb,
                                    const float* __restrict__ bnb,
                                    _Float16* __restrict__ Wf,
                                    float* __restrict__ bf,
                                    int Cout, int Cin, int Kpad) {
  int t = blockIdx.x * blockDim.x + threadIdx.x;
  int total = Cout * Kpad;
  if (t < total) {
    int o = t / Kpad, k = t - o * Kpad;
    float s = g ? g[o] * BN_SCALE : 1.0f;
    Wf[t] = (k < Cin) ? (_Float16)(W[(size_t)o * Cin + k] * s) : (_Float16)0.0f;
  }
  if (t < Cout) {
    float s = g ? g[t] * BN_SCALE : 1.0f;
    bf[t] = (lb ? s * lb[t] : 0.0f) + (bnb ? bnb[t] : 0.0f);
  }
}

// ---------------------------------------------------------------------------
// Farthest point sampling: one block per batch, min-dist array in LDS.
// ---------------------------------------------------------------------------
__global__ void fps_kernel(const float* __restrict__ xyz, int N, int stride,
                           int npoint, int* __restrict__ outIdx) {
  __shared__ float minD[4096];
  __shared__ float redv[256];
  __shared__ int   redi[256];
  __shared__ float lastp[3];
  __shared__ int   lastS;
  int b = blockIdx.x, tid = threadIdx.x;
  const float* px = xyz + (size_t)b * N * stride;
  for (int i = tid; i < N; i += 256) minD[i] = 1e10f;
  int last = 0;
  __syncthreads();
  for (int s = 0; s < npoint; ++s) {
    if (tid == 0) {
      outIdx[(size_t)b * npoint + s] = last;
      lastp[0] = px[(size_t)last * stride + 0];
      lastp[1] = px[(size_t)last * stride + 1];
      lastp[2] = px[(size_t)last * stride + 2];
    }
    __syncthreads();
    float lx = lastp[0], ly = lastp[1], lz = lastp[2];
    float bv = -1.0f; int bi = 0;
    for (int i = tid; i < N; i += 256) {
      float dx = px[(size_t)i * stride + 0] - lx;
      float dy = px[(size_t)i * stride + 1] - ly;
      float dz = px[(size_t)i * stride + 2] - lz;
      float d = dx * dx + dy * dy + dz * dz;
      float m = fminf(minD[i], d);
      minD[i] = m;
      if (m > bv) { bv = m; bi = i; }
    }
    redv[tid] = bv; redi[tid] = bi;
    __syncthreads();
    for (int off = 128; off > 0; off >>= 1) {
      if (tid < off) {
        if (redv[tid + off] > redv[tid] ||
            (redv[tid + off] == redv[tid] && redi[tid + off] < redi[tid])) {
          redv[tid] = redv[tid + off]; redi[tid] = redi[tid + off];
        }
      }
      __syncthreads();
    }
    if (tid == 0) lastS = redi[0];
    __syncthreads();
    last = lastS;
  }
}

__global__ void gather_xyz_kernel(const float* __restrict__ src, int N, int stride,
                                  const int* __restrict__ idx, int npoint,
                                  int count, float* __restrict__ dst) {
  int t = blockIdx.x * blockDim.x + threadIdx.x;
  if (t >= count * 3) return;
  int i = t / 3, c = t - i * 3;
  int b = i / npoint;
  dst[t] = src[((size_t)b * N + idx[i]) * stride + c];
}

// ---------------------------------------------------------------------------
// Ball query: first nsample points (in index order) within radius; pad w/first.
// ---------------------------------------------------------------------------
__global__ void ball_query_kernel(const float* __restrict__ xyz, int N, int stride,
                                  const float* __restrict__ newxyz,
                                  int total, int S, int ns, float r2,
                                  int* __restrict__ idx) {
  int t = blockIdx.x * blockDim.x + threadIdx.x;
  if (t >= total) return;
  int b = t / S;
  float cx = newxyz[(size_t)t * 3 + 0];
  float cy = newxyz[(size_t)t * 3 + 1];
  float cz = newxyz[(size_t)t * 3 + 2];
  const float* px = xyz + (size_t)b * N * stride;
  int cnt = 0, first = 0; bool any = false;
  for (int n = 0; n < N; ++n) {
    float dx = px[(size_t)n * stride + 0] - cx;
    float dy = px[(size_t)n * stride + 1] - cy;
    float dz = px[(size_t)n * stride + 2] - cz;
    float d2 = dx * dx + dy * dy + dz * dz;
    if (d2 <= r2) {
      if (!any) { any = true; first = n; }
      idx[(size_t)t * ns + cnt] = n;
      if (++cnt == ns) break;
    }
  }
  int fill = any ? first : 0;
  for (; cnt < ns; ++cnt) idx[(size_t)t * ns + cnt] = fill;
}

// ---------------------------------------------------------------------------
// WMMA fragment helpers (wave32, v_wmma_f32_16x16x32_f16).
// A 16x32 f16: lane<16 -> K {0..15}, lane>=16 -> K {16..31}
// B 32x16 f16: lane holds column n=lane&15, 16 contiguous K per half.
// C/D f32: element v of lane -> row v + (lane>=16)*8, col lane&15.
// ---------------------------------------------------------------------------
__device__ __forceinline__ v16h load_a_frag(const _Float16* __restrict__ arow,
                                            int half) {
  return *(const v16h*)(arow + half * 16);
}

// ---------------------------------------------------------------------------
// Fused MLP layer with async-staged weights.
// Each iteration stages the 32 weight rows (two 16-col n-tiles) used by the
// npart=0 and npart=1 wave pairs into LDS via global_load_async_to_lds_b128,
// so the rg=0/rg=1 wave pair shares one fetch of each weight row.
// finalMax: max-pool epilogue with atomicMax instead of f16 store.
// ---------------------------------------------------------------------------
__device__ __forceinline__ void wmma_layer_async(
    const _Float16* __restrict__ Ain, int Kin,
    const _Float16* __restrict__ Wf, const float* __restrict__ bias, int H,
    _Float16* __restrict__ Out, _Float16* __restrict__ WsBuf,
    int rg, int npart, int tid, int lane,
    bool finalMax, float* __restrict__ outRow) {
  int m = lane & 15, half = lane >> 4;
  int iters = H >> 5;                      // ntiles/2, H is a multiple of 32
  int chunksPerRow = Kin >> 3;             // 16B chunks per weight row
  int totalChunks = 32 * chunksPerRow;     // always a multiple of 128
  for (int it = 0; it < iters; ++it) {
    int nbase = it << 5;                   // first of 32 weight rows this iter
    __syncthreads();                       // prev iteration done reading WsBuf
    for (int c = tid; c < totalChunks; c += 128) {
      int r = c / chunksPerRow, o = c - r * chunksPerRow;
      unsigned lds_off = (unsigned)(uintptr_t)(WsBuf + (size_t)r * Kin + o * 8);
      const void* gp = (const void*)(Wf + (size_t)(nbase + r) * Kin + o * 8);
      async_load_lds_b128(lds_off, gp);
    }
    wait_asynccnt0();                      // this wave's transfers landed in LDS
    __syncthreads();                       // all waves' transfers visible

    int n0 = (nbase + npart * 16);
    int lr = npart * 16 + m;               // local weight row in WsBuf
    v8f c = {};
    for (int k0 = 0; k0 < Kin; k0 += 32) {
      const _Float16* arow = Ain + (size_t)(rg * 16 + m) * Kin + k0;
      v16h a = load_a_frag(arow, half);
      v16h bf = *(const v16h*)(WsBuf + (size_t)lr * Kin + k0 + half * 16);
      c = __builtin_amdgcn_wmma_f32_16x16x32_f16(false, a, false, bf,
                                                 (short)0, c, false, false);
    }
    float bn = bias[n0 + m];
    if (!finalMax) {
#pragma unroll
      for (int v = 0; v < 8; ++v) {
        float x = fmaxf(c[v] + bn, 0.0f);
        Out[(size_t)(rg * 16 + v + half * 8) * H + n0 + m] = (_Float16)x;
      }
    } else {
      float mx = 0.0f;  // post-ReLU values are >= 0
#pragma unroll
      for (int v = 0; v < 8; ++v) mx = fmaxf(mx, fmaxf(c[v] + bn, 0.0f));
      mx = fmaxf(mx, __shfl_xor(mx, 16, 32));  // combine rows 0-7 / 8-15
      if (lane < 16)
        atomicMax((int*)(outRow + n0 + m), __float_as_int(mx));
    }
  }
}

// ---------------------------------------------------------------------------
// Fused: gather(ball idx) + f16 convert + 3 WMMA MLP layers in LDS + max-pool.
// Block = 128 threads (4 waves), tile = 32 rows (2 x 16-row groups).
// ---------------------------------------------------------------------------
__global__ void fused_group_mlp3(const float* __restrict__ xyz, int sx,
                                 const float* __restrict__ feats, int sf, int Cf,
                                 const float* __restrict__ newxyz,
                                 const int* __restrict__ idx,
                                 int ns, int S, int N,
                                 int Kpad, int h1, int h2, int h3,
                                 const _Float16* __restrict__ W1, const float* __restrict__ b1,
                                 const _Float16* __restrict__ W2, const float* __restrict__ b2,
                                 const _Float16* __restrict__ W3, const float* __restrict__ b3,
                                 float* __restrict__ out, int outStride, int outOff) {
  __shared__ _Float16 Xs[32 * 352];
  __shared__ _Float16 Ys[32 * 256];
  __shared__ _Float16 Ws[32 * 352];   // async-staged weight rows
  __shared__ int   srcS[32];
  __shared__ int   rowBS[32];
  __shared__ float ctrS[32][3];
  int tid = threadIdx.x;
  int lane = tid & 31, wave = tid >> 5;
  int rg = wave & 1, npart = wave >> 1;
  int rows0 = blockIdx.x * 32;

  if (tid < 32) {
    int grow = rows0 + tid;
    int g = grow / ns, j = grow - g * ns;
    srcS[tid] = idx[(size_t)g * ns + j];
    rowBS[tid] = g / S;
    ctrS[tid][0] = newxyz[(size_t)g * 3 + 0];
    ctrS[tid][1] = newxyz[(size_t)g * 3 + 1];
    ctrS[tid][2] = newxyz[(size_t)g * 3 + 2];
  }
  __syncthreads();

  // Stage gathered group rows into LDS as f16, K zero-padded.
  for (int t = tid; t < 32 * Kpad; t += 128) {
    int r = t / Kpad, k = t - r * Kpad;
    int src = srcS[r], b = rowBS[r];
    float v;
    if (k < 3)            v = xyz[((size_t)b * N + src) * sx + k] - ctrS[r][k];
    else if (k < 3 + Cf)  v = feats[((size_t)b * N + src) * sf + (k - 3)];
    else                  v = 0.0f;
    Xs[(size_t)r * Kpad + k] = (_Float16)v;
  }
  __syncthreads();

  wmma_layer_async(Xs, Kpad, W1, b1, h1, Ys, Ws, rg, npart, tid, lane,
                   false, nullptr);
  __syncthreads();
  wmma_layer_async(Ys, h1, W2, b2, h2, Xs, Ws, rg, npart, tid, lane,
                   false, nullptr);
  __syncthreads();
  int g_rg = (rows0 + rg * 16) / ns;
  wmma_layer_async(Xs, h2, W3, b3, h3, nullptr, Ws, rg, npart, tid, lane,
                   true, out + (size_t)g_rg * outStride + outOff);
}

// ---------------------------------------------------------------------------
// Generic WMMA GEMM  C = act(A[M,K] @ Wf[H,K]^T + bias).
// mode 0: f16 store; mode 1: atomicMax f32 pooled over `pool` rows; mode 2: sigmoid f32.
// ---------------------------------------------------------------------------
__global__ void gemm_wmma(const _Float16* __restrict__ A, int M, int K,
                          const _Float16* __restrict__ Wf,
                          const float* __restrict__ bias, int H,
                          int mode, int pool,
                          _Float16* __restrict__ outH,
                          float* __restrict__ outF) {
  int tid = threadIdx.x;
  int lane = tid & 31, wave = tid >> 5;
  int m = lane & 15, half = lane >> 4;
  int rows0 = blockIdx.x * 16;
  int ntiles = (H + 15) >> 4;
  int nt = blockIdx.y * 4 + wave;
  if (nt >= ntiles) return;
  int n0 = nt << 4;

  v8f c = {};
  for (int k0 = 0; k0 < K; k0 += 32) {
    const _Float16* arow = A + (size_t)(rows0 + m) * K + k0;
    __builtin_prefetch(arow + 64, 0, 1);
    v16h a = load_a_frag(arow, half);
    int wr = n0 + m; if (wr >= H) wr = H - 1;  // clamp for tiny H (stores guarded)
    v16h bf = *(const v16h*)(Wf + (size_t)wr * K + k0 + half * 16);
    c = __builtin_amdgcn_wmma_f32_16x16x32_f16(false, a, false, bf,
                                               (short)0, c, false, false);
  }
  int n = n0 + m;
  float bn = (n < H) ? bias[n] : 0.0f;
  if (mode == 0) {
#pragma unroll
    for (int v = 0; v < 8; ++v) {
      float x = fmaxf(c[v] + bn, 0.0f);
      if (n < H) outH[(size_t)(rows0 + v + half * 8) * H + n] = (_Float16)x;
    }
  } else if (mode == 1) {
    float mx = 0.0f;
#pragma unroll
    for (int v = 0; v < 8; ++v) mx = fmaxf(mx, fmaxf(c[v] + bn, 0.0f));
    mx = fmaxf(mx, __shfl_xor(mx, 16, 32));
    if (lane < 16 && n < H)
      atomicMax((int*)&outF[(size_t)(rows0 / pool) * H + n], __float_as_int(mx));
  } else {
#pragma unroll
    for (int v = 0; v < 8; ++v) {
      float x = c[v] + bn;
      if (n < H) outF[(size_t)(rows0 + v + half * 8) * H + n] = 1.0f / (1.0f + expf(-x));
    }
  }
}

// SA3 group-all input: [16*128, 672] f16 = concat(xyz2, f2) zero-padded.
__global__ void build_groupall_kernel(const float* __restrict__ xyz2,
                                      const float* __restrict__ f2,
                                      _Float16* __restrict__ Xg) {
  int t = blockIdx.x * blockDim.x + threadIdx.x;
  if (t >= 2048 * 672) return;
  int r = t / 672, k = t - r * 672;
  float v;
  if (k < 3)        v = xyz2[(size_t)r * 3 + k];
  else if (k < 643) v = f2[(size_t)r * 640 + (k - 3)];
  else              v = 0.0f;
  Xg[t] = (_Float16)v;
}

__global__ void cvt_f32_f16_kernel(const float* __restrict__ src,
                                   _Float16* __restrict__ dst, int n) {
  int t = blockIdx.x * blockDim.x + threadIdx.x;
  if (t < n) dst[t] = (_Float16)src[t];
}

// ---------------------------------------------------------------------------
// Host orchestration
// ---------------------------------------------------------------------------
static inline char* wsAlloc(char*& cur, size_t bytes) {
  char* p = cur;
  cur += (bytes + 255) & ~(size_t)255;
  return p;
}
static inline int roundup32(int x) { return (x + 31) & ~31; }

extern "C" void kernel_launch(void* const* d_in, const int* in_sizes, int n_in,
                              void* d_out, int out_size, void* d_ws, size_t ws_size,
                              hipStream_t stream) {
  (void)in_sizes; (void)n_in; (void)out_size; (void)ws_size;
  const int B = 16, N = 4096;

  // ---- parse inputs (setup_inputs dict order, depth-first insertion order)
  int ci = 0;
  const float* pc = (const float*)d_in[ci++];  // pointcloud [16,4096,19]
  const float *saW[21], *saG[21], *saB[21];    // sa1(9) sa2(9) sa3(3): W,g,b each
  for (int l = 0; l < 21; ++l) {
    saW[l] = (const float*)d_in[ci++];
    saG[l] = (const float*)d_in[ci++];
    saB[l] = (const float*)d_in[ci++];
  }
  const float* l1W = (const float*)d_in[ci++]; const float* l1b = (const float*)d_in[ci++];
  const float* bn1g = (const float*)d_in[ci++]; const float* bn1b = (const float*)d_in[ci++];
  const float* l2W = (const float*)d_in[ci++]; const float* l2b = (const float*)d_in[ci++];
  const float* bn2g = (const float*)d_in[ci++]; const float* bn2b = (const float*)d_in[ci++];
  const float* l3W = (const float*)d_in[ci++]; const float* l3b = (const float*)d_in[ci++];
  const float* bn3g = (const float*)d_in[ci++]; const float* bn3b = (const float*)d_in[ci++];
  const float* l4W = (const float*)d_in[ci++]; const float* l4b = (const float*)d_in[ci++];

  static const int saCin[21]  = {22,32,32, 22,64,64, 22,64,64,
                                 323,64,64, 323,128,128, 323,128,128,
                                 643,256,512};
  static const int saCout[21] = {32,32,64, 64,64,128, 64,64,128,
                                 64,64,128, 128,128,256, 128,128,256,
                                 256,512,1024};

  char* cur = (char*)d_ws;
  auto fold = [&](const float* W, const float* g, const float* lb, const float* bnb,
                  int cout, int cin, _Float16*& wf, float*& bf) {
    int kpad = roundup32(cin);
    wf = (_Float16*)wsAlloc(cur, (size_t)cout * kpad * sizeof(_Float16));
    bf = (float*)wsAlloc(cur, (size_t)cout * sizeof(float));
    int total = cout * kpad;
    fold_weights_kernel<<<(total + 255) / 256, 256, 0, stream>>>(
        W, g, lb, bnb, wf, bf, cout, cin, kpad);
  };

  _Float16* saWf[21]; float* saBf[21];
  for (int l = 0; l < 21; ++l)
    fold(saW[l], saG[l], nullptr, saB[l], saCout[l], saCin[l], saWf[l], saBf[l]);
  _Float16 *cw1, *cw2, *cw3, *cw4; float *cb1, *cb2, *cb3, *cb4;
  fold(l1W, bn1g, l1b, bn1b, 512, 1024, cw1, cb1);
  fold(l2W, bn2g, l2b, bn2b, 256, 512,  cw2, cb2);
  fold(l3W, bn3g, l3b, bn3b, 128, 256,  cw3, cb3);
  fold(l4W, nullptr, l4b, nullptr, 2, 128, cw4, cb4);

  // ---- workspace buffers
  int*   idx1   = (int*)wsAlloc(cur, (size_t)B * 512 * sizeof(int));
  int*   idx2   = (int*)wsAlloc(cur, (size_t)B * 128 * sizeof(int));
  float* xyz1   = (float*)wsAlloc(cur, (size_t)B * 512 * 3 * sizeof(float));
  float* xyz2   = (float*)wsAlloc(cur, (size_t)B * 128 * 3 * sizeof(float));
  int*   ballIdx= (int*)wsAlloc(cur, (size_t)B * 512 * 128 * sizeof(int));   // reused per branch
  float* f1     = (float*)wsAlloc(cur, (size_t)B * 512 * 320 * sizeof(float));
  float* f2     = (float*)wsAlloc(cur, (size_t)B * 128 * 640 * sizeof(float));
  _Float16* Xg   = (_Float16*)wsAlloc(cur, (size_t)2048 * 672 * sizeof(_Float16));
  _Float16* act1 = (_Float16*)wsAlloc(cur, (size_t)2048 * 256 * sizeof(_Float16));
  _Float16* act2 = (_Float16*)wsAlloc(cur, (size_t)2048 * 512 * sizeof(_Float16));
  float* pooled  = (float*)wsAlloc(cur, (size_t)16 * 1024 * sizeof(float));
  _Float16* pooledh = (_Float16*)wsAlloc(cur, (size_t)16 * 1024 * sizeof(_Float16));
  _Float16* c1 = (_Float16*)wsAlloc(cur, (size_t)16 * 512 * sizeof(_Float16));
  _Float16* c2 = (_Float16*)wsAlloc(cur, (size_t)16 * 256 * sizeof(_Float16));
  _Float16* c3 = (_Float16*)wsAlloc(cur, (size_t)16 * 128 * sizeof(_Float16));

  hipMemsetAsync(f1, 0, (size_t)B * 512 * 320 * sizeof(float), stream);
  hipMemsetAsync(f2, 0, (size_t)B * 128 * 640 * sizeof(float), stream);
  hipMemsetAsync(pooled, 0, (size_t)16 * 1024 * sizeof(float), stream);

  // ---- SA1: FPS 4096 -> 512 centers
  fps_kernel<<<B, 256, 0, stream>>>(pc, N, 19, 512, idx1);
  gather_xyz_kernel<<<(B * 512 * 3 + 255) / 256, 256, 0, stream>>>(
      pc, N, 19, idx1, 512, B * 512, xyz1);

  {  // SA1 branches
    const float radii[3] = {0.1f, 0.2f, 0.4f};
    const int nsamp[3] = {16, 32, 128};
    const int off[3] = {0, 64, 192};
    for (int br = 0; br < 3; ++br) {
      int ns = nsamp[br], S = 512;
      int total = B * S;
      ball_query_kernel<<<(total + 255) / 256, 256, 0, stream>>>(
          pc, N, 19, xyz1, total, S, ns, radii[br] * radii[br], ballIdx);
      int li = br * 3;  // layer base in sa table
      size_t rows = (size_t)total * ns;
      fused_group_mlp3<<<(int)(rows / 32), 128, 0, stream>>>(
          pc, 19, pc, 19, 19, xyz1, ballIdx, ns, S, N,
          32, saCout[li], saCout[li + 1], saCout[li + 2],
          saWf[li], saBf[li], saWf[li + 1], saBf[li + 1], saWf[li + 2], saBf[li + 2],
          f1, 320, off[br]);
    }
  }

  // ---- SA2: FPS 512 -> 128 centers
  fps_kernel<<<B, 256, 0, stream>>>(xyz1, 512, 3, 128, idx2);
  gather_xyz_kernel<<<(B * 128 * 3 + 255) / 256, 256, 0, stream>>>(
      xyz1, 512, 3, idx2, 128, B * 128, xyz2);

  {  // SA2 branches
    const float radii[3] = {0.2f, 0.4f, 0.8f};
    const int nsamp[3] = {32, 64, 128};
    const int off[3] = {0, 128, 384};
    for (int br = 0; br < 3; ++br) {
      int ns = nsamp[br], S = 128;
      int total = B * S;
      ball_query_kernel<<<(total + 255) / 256, 256, 0, stream>>>(
          xyz1, 512, 3, xyz2, total, S, ns, radii[br] * radii[br], ballIdx);
      int li = 9 + br * 3;
      size_t rows = (size_t)total * ns;
      fused_group_mlp3<<<(int)(rows / 32), 128, 0, stream>>>(
          xyz1, 3, f1, 320, 320, xyz2, ballIdx, ns, S, 512,
          352, saCout[li], saCout[li + 1], saCout[li + 2],
          saWf[li], saBf[li], saWf[li + 1], saBf[li + 1], saWf[li + 2], saBf[li + 2],
          f2, 640, off[br]);
    }
  }

  // ---- SA3: group-all [2048, 672] -> 256 -> 512 -> 1024, max over 128 pts
  build_groupall_kernel<<<(2048 * 672 + 255) / 256, 256, 0, stream>>>(xyz2, f2, Xg);
  {
    dim3 g1(2048 / 16, (256 / 16 + 3) / 4);
    gemm_wmma<<<g1, 128, 0, stream>>>(Xg, 2048, 672, saWf[18], saBf[18], 256,
                                      0, 0, act1, nullptr);
    dim3 g2(2048 / 16, (512 / 16 + 3) / 4);
    gemm_wmma<<<g2, 128, 0, stream>>>(act1, 2048, 256, saWf[19], saBf[19], 512,
                                      0, 0, act2, nullptr);
    dim3 g3(2048 / 16, (1024 / 16 + 3) / 4);
    gemm_wmma<<<g3, 128, 0, stream>>>(act2, 2048, 512, saWf[20], saBf[20], 1024,
                                      1, 128, nullptr, pooled);
  }
  cvt_f32_f16_kernel<<<(16 * 1024 + 255) / 256, 256, 0, stream>>>(pooled, pooledh, 16 * 1024);

  // ---- classifier head: 1024 -> 512 -> 256 -> 128 -> 2 (sigmoid)
  {
    dim3 g1(1, (512 / 16 + 3) / 4);
    gemm_wmma<<<g1, 128, 0, stream>>>(pooledh, 16, 1024, cw1, cb1, 512, 0, 0, c1, nullptr);
    dim3 g2(1, (256 / 16 + 3) / 4);
    gemm_wmma<<<g2, 128, 0, stream>>>(c1, 16, 512, cw2, cb2, 256, 0, 0, c2, nullptr);
    dim3 g3(1, (128 / 16 + 3) / 4);
    gemm_wmma<<<g3, 128, 0, stream>>>(c2, 16, 256, cw3, cb3, 128, 0, 0, c3, nullptr);
    dim3 g4(1, 1);
    gemm_wmma<<<g4, 128, 0, stream>>>(c3, 16, 128, cw4, cb4, 2, 2, 0, nullptr, (float*)d_out);
  }
}
